// SelfAttentionHead_20426864460214
// MI455X (gfx1250) — compile-verified, matrix operand order checked by reading, not run on hardware
//
#include <hip/hip_runtime.h>
#include <hip/hip_bf16.h>

typedef __bf16 bf16_t;
typedef __attribute__((ext_vector_type(16))) __bf16 v16bf;
typedef __attribute__((ext_vector_type(8)))  float  v8f;
typedef __attribute__((ext_vector_type(4)))  unsigned v4u;
typedef __attribute__((ext_vector_type(8)))  int      v8i;
typedef __attribute__((ext_vector_type(4)))  int      v4i;

#define EMB   1024
#define HS    64
#define SEQ   4096
#define BATCH 8
#define NROW  (BATCH * SEQ)          // 32768 total rows
#define WELEM (HS * EMB)             // 65536 elems per weight matrix
#define WROWS 192                    // 3 projections x 64 output cols
#define KSTEP 32                     // contraction step per WMMA
#define SLICE (WROWS * KSTEP)        // bf16 elems per staged weight slice (12KB)

// native RNE float -> bf16 (v_cvt_pk_bf16_f32 when pairs fuse)
__device__ __forceinline__ bf16_t f2bf(float f) { return (bf16_t)f; }

// K index for VGPR pair v of a 16-bit A/B fragment (ISA 7.12.2):
// lanes 0-15 hold K={0..7,16..23}, lanes 16-31 hold K={8..15,24..31}, 2 per VGPR
__device__ __forceinline__ int kofv(int v, int half) {
    return (v < 4) ? (half * 8 + 2 * v) : (16 + half * 8 + 2 * (v - 4));
}

// load a bf16 A/B fragment whose per-lane data is at p[kofv(v)] (pairs contiguous)
__device__ __forceinline__ v16bf ldfrag(const bf16_t* __restrict__ p, int half) {
    v16bf f;
#pragma unroll
    for (int v = 0; v < 8; ++v) {
        int k = kofv(v, half);
        f[2 * v]     = p[k];
        f[2 * v + 1] = p[k + 1];
    }
    return f;
}

// TDM: DMA a [WROWS x KSTEP] bf16 tile (row stride EMB) from global into LDS.
// D# per CDNA5 ISA: group0 = {count/lds_addr/global_addr/type},
// group1 = {data_size, tensor_dim0/1, tile_dim0/1, tensor_dim0_stride}.
// amdgpu-toolchain (clang-23) builtin arity: 6 args, fifth is int32x8.
__device__ __forceinline__ void tdm_load_wslice(const bf16_t* gsrc,
                                                unsigned lds_off) {
    unsigned long long ga = (unsigned long long)(size_t)gsrc;
    v4u g0;
    g0[0] = 1u;                                        // count=1, user mode
    g0[1] = lds_off;                                   // lds_addr (bytes)
    g0[2] = (unsigned)(ga & 0xFFFFFFFFu);              // global_addr[31:0]
    g0[3] = (unsigned)((ga >> 32) & 0x1FFFFFFu)        // global_addr[56:32]
            | (2u << 30);                              // type=2 ("image")
    v8i g1;
    g1[0] = (int)(1u << 16);        // workgroup_mask=0, data_size=1 (2 bytes)
    g1[1] = (int)((unsigned)EMB << 16);   // tensor_dim0[15:0] in bits[31:16]
    g1[2] = (int)((unsigned)WROWS << 16); // tensor_dim0 hi=0; tensor_dim1 lo16
    g1[3] = (int)((unsigned)KSTEP << 16); // tensor_dim1 hi=0; tile_dim0=32
    g1[4] = (int)WROWS;                   // tile_dim1=192; tile_dim2=0
    g1[5] = (int)EMB;                     // tensor_dim0_stride low32 = 1024
    g1[6] = 0;                            // stride0 hi16; dim1_stride lo16
    g1[7] = 0;                            // dim1_stride hi32
    const v4i z4 = {0, 0, 0, 0};          // 2-D tensor: groups 2/3 unused
    const v8i z8 = {0, 0, 0, 0, 0, 0, 0, 0};
    __builtin_amdgcn_tensor_load_to_lds(g0, g1, z4, z4, z8, 0);
}

// ---------------------------------------------------------------- kernel 1
__global__ void cvt_weights(const float* __restrict__ Wq,
                            const float* __restrict__ Wk,
                            const float* __restrict__ Wv,
                            bf16_t* __restrict__ Wb) {
    int i = blockIdx.x * blockDim.x + threadIdx.x;
    if (i < WELEM) {
        Wb[i]             = f2bf(Wq[i]);
        Wb[WELEM + i]     = f2bf(Wk[i]);
        Wb[2 * WELEM + i] = f2bf(Wv[i]);
    }
}

// ---------------------------------------------------------------- kernel 2
// Fused QKV projection. One wave computes a 16-row x 64-col tile of Q, K, V.
// Weight K-slices are TDM-staged into double-buffered LDS, shared by 8 waves.
// Q,K stored row-major bf16 [NROW][64]; V stored transposed bf16 [B][64][SEQ].
__global__ void qkv_proj(const float* __restrict__ X,
                         const bf16_t* __restrict__ Wb,
                         bf16_t* __restrict__ Qb,
                         bf16_t* __restrict__ Kb,
                         bf16_t* __restrict__ Vt) {
    const int lane = threadIdx.x & 31;
    const int half = lane >> 4;
    const int r    = lane & 15;
    const int w    = threadIdx.x >> 5;          // wave in block
    const int wave = (blockIdx.x * blockDim.x + threadIdx.x) >> 5;
    const int r0   = wave * 16;                 // first row of this tile
    const int b    = r0 >> 12;                  // batch index (r0 / 4096)
    const int s0   = r0 & (SEQ - 1);

    __shared__ bf16_t wlds[2][SLICE];           // 2 x 12KB weight slices

    v8f acc[12];                                // [proj 0..2][ntile 0..3]
    const v8f vz = {0.f, 0.f, 0.f, 0.f, 0.f, 0.f, 0.f, 0.f};
#pragma unroll
    for (int i = 0; i < 12; ++i) acc[i] = vz;

    const float* xrow = X + (size_t)(r0 + r) * EMB;

    // prologue: stage slice 0
    if (w == 0)
        tdm_load_wslice(Wb, (unsigned)(size_t)(&wlds[0][0]));

    for (int e0 = 0; e0 < EMB; e0 += KSTEP) {
        const int cur = (e0 >> 5) & 1;

        // slice for this step complete; all waves done reading buffer `cur`
        // (it was last read two steps ago, before the previous barrier)
        if (w == 0) __builtin_amdgcn_s_wait_tensorcnt(0);
        __syncthreads();

        // kick off DMA of the next slice into the other buffer
        if (w == 0 && e0 + KSTEP < EMB)
            tdm_load_wslice(Wb + e0 + KSTEP,
                            (unsigned)(size_t)(&wlds[cur ^ 1][0]));

        // prefetch next A slice (global_prefetch_b8; no LOADcnt cost)
        if (e0 + KSTEP < EMB) __builtin_prefetch(xrow + e0 + KSTEP, 0, 0);

        // A fragment from fp32 input, convert to bf16 (packed cvt)
        v16bf a;
#pragma unroll
        for (int v = 0; v < 8; ++v) {
            int k = kofv(v, half);
            a[2 * v]     = f2bf(xrow[e0 + k]);
            a[2 * v + 1] = f2bf(xrow[e0 + k + 1]);
        }

        // 12 B fragments straight out of LDS (ds_load_b128 pairs)
        const bf16_t* wslice = &wlds[cur][0];
#pragma unroll
        for (int idx = 0; idx < 12; ++idx) {
            v16bf bf = ldfrag(wslice + (size_t)(idx * 16 + r) * KSTEP, half);
            acc[idx] = __builtin_amdgcn_wmma_f32_16x16x32_bf16(
                false, a, false, bf, (short)0, acc[idx], false, false);
        }
    }

    // epilogue: D layout row m = v + 8*half, col n = nt*16 + r
#pragma unroll
    for (int nt = 0; nt < 4; ++nt) {
#pragma unroll
        for (int v = 0; v < 8; ++v) {
            int m = v + 8 * half;
            int n = nt * 16 + r;
            Qb[(size_t)(r0 + m) * HS + n] = f2bf(acc[0 * 4 + nt][v]);
            Kb[(size_t)(r0 + m) * HS + n] = f2bf(acc[1 * 4 + nt][v]);
            Vt[(size_t)b * HS * SEQ + (size_t)n * SEQ + (s0 + m)] =
                f2bf(acc[2 * 4 + nt][v]);
        }
    }
}

// ---------------------------------------------------------------- kernel 3
// Flash attention: one wave per 16-query tile, causal loop over 32 keys/iter.
__global__ void flash_attn(const bf16_t* __restrict__ Qb,
                           const bf16_t* __restrict__ Kb,
                           const bf16_t* __restrict__ Vt,
                           float* __restrict__ O) {
    const int lane = threadIdx.x & 31;
    const int half = lane >> 4;
    const int r    = lane & 15;
    const int w    = threadIdx.x >> 5;            // wave in block (0..7)
    const int qt   = blockIdx.x * 8 + w;          // query tile 0..255
    const int b    = blockIdx.y;
    const int q0   = qt * 16;

    __shared__ bf16_t plds[8 * 16 * 32];          // per-wave 16x32 bf16 P tile
    bf16_t* myP = plds + w * (16 * 32);

    // preload Q fragments (two K-halves of H=64)
    const bf16_t* qrow = Qb + ((size_t)b * SEQ + q0 + r) * HS;
    v16bf qf[2];
#pragma unroll
    for (int hh = 0; hh < 2; ++hh) qf[hh] = ldfrag(qrow + hh * 32, half);

    const v8f vz = {0.f, 0.f, 0.f, 0.f, 0.f, 0.f, 0.f, 0.f};
    v8f o[4];
#pragma unroll
    for (int nt = 0; nt < 4; ++nt) o[nt] = vz;
    float mrow[8], lrow[8];
#pragma unroll
    for (int v = 0; v < 8; ++v) { mrow[v] = -1.0e30f; lrow[v] = 0.0f; }

    const float scale = 0.125f;                   // 1/sqrt(64)
    const bf16_t* Kbase  = Kb + (size_t)b * SEQ * HS;
    const bf16_t* Vtbase = Vt + (size_t)b * HS * SEQ;

    for (int k0 = 0; k0 < q0 + 16; k0 += 32) {
        // ---- scores: two 16-key column tiles (batch both K-half loads)
        v8f sc[2];
#pragma unroll
        for (int kt = 0; kt < 2; ++kt) {
            const bf16_t* krow = Kbase + (size_t)(k0 + kt * 16 + r) * HS;
            v16bf kb0 = ldfrag(krow, half);
            v16bf kb1 = ldfrag(krow + 32, half);
            v8f s = vz;
            s = __builtin_amdgcn_wmma_f32_16x16x32_bf16(
                false, qf[0], false, kb0, (short)0, s, false, false);
            s = __builtin_amdgcn_wmma_f32_16x16x32_bf16(
                false, qf[1], false, kb1, (short)0, s, false, false);
            // scale + causal mask
#pragma unroll
            for (int v = 0; v < 8; ++v) {
                int qq = q0 + v + 8 * half;
                int kk = k0 + kt * 16 + r;
                float val = s[v] * scale;
                s[v] = (kk > qq) ? -1.0e30f : val;
            }
            sc[kt] = s;
        }

        // ---- online softmax: row max across 16 lanes, then exp / rescale
        float mnew[8], corr[8];
#pragma unroll
        for (int v = 0; v < 8; ++v) {
            float mx = fmaxf(sc[0][v], sc[1][v]);
            mx = fmaxf(mx, __shfl_xor(mx, 1, 16));
            mx = fmaxf(mx, __shfl_xor(mx, 2, 16));
            mx = fmaxf(mx, __shfl_xor(mx, 4, 16));
            mx = fmaxf(mx, __shfl_xor(mx, 8, 16));
            mnew[v] = fmaxf(mrow[v], mx);
            corr[v] = __expf(mrow[v] - mnew[v]);
            mrow[v] = mnew[v];
            lrow[v] *= corr[v];
        }
#pragma unroll
        for (int nt = 0; nt < 4; ++nt)
#pragma unroll
            for (int v = 0; v < 8; ++v) o[nt][v] *= corr[v];

        // ---- P = exp(s - m), write D-layout tile to LDS, accumulate l
#pragma unroll
        for (int v = 0; v < 8; ++v) {
            float p0 = __expf(sc[0][v] - mnew[v]);
            float p1 = __expf(sc[1][v] - mnew[v]);
            int m = v + 8 * half;
            myP[m * 32 + r]      = f2bf(p0);
            myP[m * 32 + 16 + r] = f2bf(p1);
            float ps = p0 + p1;
            ps += __shfl_xor(ps, 1, 16);
            ps += __shfl_xor(ps, 2, 16);
            ps += __shfl_xor(ps, 4, 16);
            ps += __shfl_xor(ps, 8, 16);
            lrow[v] += ps;
        }

        // wave-private LDS: wait own DS stores before transposed reload
        asm volatile("s_wait_dscnt 0x0" ::: "memory");

        // ---- reload P as A-fragment (transpose via LDS)
        v16bf pf = ldfrag(myP + r * 32, half);

        // ---- O += P x V: batch all four V fragments, then WMMA group
        v16bf vb[4];
#pragma unroll
        for (int nt = 0; nt < 4; ++nt)
            vb[nt] = ldfrag(Vtbase + (size_t)(nt * 16 + r) * SEQ + k0, half);
#pragma unroll
        for (int nt = 0; nt < 4; ++nt)
            o[nt] = __builtin_amdgcn_wmma_f32_16x16x32_bf16(
                false, pf, false, vb[nt], (short)0, o[nt], false, false);
    }

    // ---- finalize: divide by l, store fp32
#pragma unroll
    for (int nt = 0; nt < 4; ++nt) {
#pragma unroll
        for (int v = 0; v < 8; ++v) {
            int m = v + 8 * half;
            float inv = 1.0f / lrow[v];
            O[((size_t)b * SEQ + q0 + m) * HS + nt * 16 + r] = o[nt][v] * inv;
        }
    }
}

// ---------------------------------------------------------------- launcher
extern "C" void kernel_launch(void* const* d_in, const int* in_sizes, int n_in,
                              void* d_out, int out_size, void* d_ws, size_t ws_size,
                              hipStream_t stream) {
    const float* X  = (const float*)d_in[0];
    const float* Wq = (const float*)d_in[1];
    const float* Wk = (const float*)d_in[2];
    const float* Wv = (const float*)d_in[3];
    float* O = (float*)d_out;

    // workspace layout (bf16 elements)
    bf16_t* Wb = (bf16_t*)d_ws;                  // 3 * 65536
    bf16_t* Qb = Wb + 3 * WELEM;                 // 32768 * 64
    bf16_t* Kb = Qb + (size_t)NROW * HS;         // 32768 * 64
    bf16_t* Vt = Kb + (size_t)NROW * HS;         // 8 * 64 * 4096

    cvt_weights<<<(WELEM + 255) / 256, 256, 0, stream>>>(Wq, Wk, Wv, Wb);

    // 32768 rows / 16 per wave = 2048 waves; 8 waves per block
    qkv_proj<<<256, 256, 0, stream>>>(X, Wb, Qb, Kb, Vt);

    // 256 query tiles per batch, 8 waves per block
    flash_attn<<<dim3(32, BATCH), 256, 0, stream>>>(Qb, Kb, Vt, O);
}
